// ARCategoricalPolicy_31112743092606
// MI455X (gfx1250) — matrix operand order repaired and checked by reading, not compiled, not run on hardware
//
#include <hip/hip_runtime.h>
#include <hip/hip_bf16.h>
#include <stdint.h>

// ---------------------------------------------------------------------------
// AR LSTM sampler for MI455X (gfx1250), wave32 + WMMA bf16.
//   B=16384, H=512, A=64, L=256.  Each 512-thread block (16 wave32) owns a
//   32-row batch tile for the whole sequence.  c-state lives in VGPRs, h is
//   staged in dynamic LDS as WMMA A-fragments (bf16), weights pre-converted
//   once to bf16 B-fragments in workspace (L2 resident).  Each B fragment is
//   reused across 2 M-tiles -> 32 FLOP per L2 byte of weight traffic.
// ---------------------------------------------------------------------------

#define HID    512
#define ALPH   64
#define SEQL   256
#define BATCHN 16384
#define TILE_B 32
#define BLOCK  512
#define NWAVE  16

// dynamic LDS layout:
//   Ast: 2 * 32 * 32 * 16 bf16  (A fragments, [mt][kt][lane][j])  = 65536 B
//   logits: 32 * 64 f32                                          =  8192 B
//   toks: 32 i32                                                 =   128 B
#define SMEM_AST_BYTES   (2 * 32 * 32 * 16 * 2)
#define SMEM_LOG_BYTES   (TILE_B * ALPH * 4)
#define SMEM_TOTAL       (SMEM_AST_BYTES + SMEM_LOG_BYTES + TILE_B * 4)

typedef __attribute__((ext_vector_type(16))) __bf16 v16bf;
typedef __attribute__((ext_vector_type(8)))  float  v8f;

__device__ __forceinline__ float fsig(float x) { return 1.0f / (1.0f + __expf(-x)); }
__device__ __forceinline__ float ftanh(float x) {
    float e = __expf(2.0f * x);
    return (e - 1.0f) / (e + 1.0f);
}
__device__ __forceinline__ uint32_t mix32(uint32_t x) {
    x ^= x >> 16; x *= 0x7feb352du;
    x ^= x >> 15; x *= 0x846ca68bu;
    x ^= x >> 16; return x;
}

// ---------------------------------------------------------------------------
// Weight conversion: f32 (4H,H) row-major -> bf16 combined W^T (K=1024,N=2048)
// as WMMA B-fragments: frag (kt,ntile) = 512 bf16; lane l holds 16 contiguous
// bf16 (32B) = {k = (j>=8?16:0) + (l>=16?8:0) + (j&7), n = ntile*16 + (l&15)}.
// ---------------------------------------------------------------------------
__global__ __launch_bounds__(256) void conv_wc_kernel(
    const float* __restrict__ w_ih, const float* __restrict__ w_hh,
    unsigned short* __restrict__ wc_raw)
{
    int idx = blockIdx.x * 256 + threadIdx.x;             // 0 .. 1024*2048-1
    if (idx >= 1024 * 2048) return;
    int k = idx >> 11;                                    // 0..1023
    int n = idx & 2047;                                   // 0..2047
    float v = (k < HID) ? w_ih[(size_t)n * HID + k]
                        : w_hh[(size_t)n * HID + (k - HID)];
    int kt = k >> 5, kk = k & 31, ntile = n >> 4, nn = n & 15;
    int laneB = ((kk >> 3) & 1) * 16 + nn;
    int j = ((kk >> 4) & 1) * 8 + (kk & 7);
    size_t off = ((size_t)(kt * 128 + ntile) * 512) + (size_t)laneB * 16 + j;
    ((__bf16*)wc_raw)[off] = (__bf16)v;
}

// head_w (A,H) f32 -> bf16 W^T (K=512,N=64) in B-fragment layout (4 ntiles).
__global__ __launch_bounds__(256) void conv_hw_kernel(
    const float* __restrict__ head_w, unsigned short* __restrict__ hw_raw)
{
    int idx = blockIdx.x * 256 + threadIdx.x;             // 0 .. 512*64-1
    if (idx >= HID * ALPH) return;
    int k = idx >> 6;
    int n = idx & 63;
    float v = head_w[(size_t)n * HID + k];
    int kt = k >> 5, kk = k & 31, ntile = n >> 4, nn = n & 15;
    int laneB = ((kk >> 3) & 1) * 16 + nn;
    int j = ((kk >> 4) & 1) * 8 + (kk & 7);
    size_t off = ((size_t)(kt * 4 + ntile) * 512) + (size_t)laneB * 16 + j;
    ((__bf16*)hw_raw)[off] = (__bf16)v;
}

// combined bias bc[n] = b_ih[n] + b_hh[n]
__global__ __launch_bounds__(256) void conv_bias_kernel(
    const float* __restrict__ b_ih, const float* __restrict__ b_hh,
    float* __restrict__ bc)
{
    int idx = blockIdx.x * 256 + threadIdx.x;
    if (idx < 4 * HID) bc[idx] = b_ih[idx] + b_hh[idx];
}

// ---------------------------------------------------------------------------
// Main persistent-per-tile kernel.
// ---------------------------------------------------------------------------
__global__ __launch_bounds__(BLOCK) void lstm_sample_kernel(
    const float* __restrict__ embed,             // (A+1, H) f32
    const unsigned short* __restrict__ wc_raw,   // bf16 B-frags (1024x2048)
    const unsigned short* __restrict__ hw_raw,   // bf16 B-frags (512x64)
    const float* __restrict__ bc,                // (4H,) combined bias
    const float* __restrict__ head_b,            // (A,)
    int*   __restrict__ out_x,                   // (B, L) int32
    float* __restrict__ out_logp)                // (B,)
{
    const __bf16* Wc = (const __bf16*)wc_raw;
    const __bf16* Hw = (const __bf16*)hw_raw;

    extern __shared__ __align__(32) char smem[];
    __bf16* Ast      = (__bf16*)smem;                      // [mt][kt][lane][16]
    float*  logits_s = (float*)(smem + SMEM_AST_BYTES);    // [row][64]
    int*    toks     = (int*)(smem + SMEM_AST_BYTES + SMEM_LOG_BYTES);

    const int tid   = threadIdx.x;
    const int wave  = tid >> 5;                  // 0..15
    const int lane  = tid & 31;
    const int nn    = lane & 15;
    const int mbase = (lane < 16) ? 0 : 8;       // C/D layout: lanes 16-31 = M+8

    // Wave w owns gate-columns [w*32, w*32+32) of each of the 4 gates,
    // for both M-tiles (rows 0-15 and 16-31).
    v8f acc[4][2][2];        // [gate][ct][mt] 16x16 f32 accum tiles
    v8f creg[2][2];          // c-state [ct][mt]
    #pragma unroll
    for (int ct = 0; ct < 2; ++ct)
        #pragma unroll
        for (int mt = 0; mt < 2; ++mt) { v8f z = {}; creg[ct][mt] = z; }

    float biasr[4][2];
    #pragma unroll
    for (int g = 0; g < 4; ++g)
        #pragma unroll
        for (int ct = 0; ct < 2; ++ct)
            biasr[g][ct] = bc[g * HID + wave * 32 + ct * 16 + nn];
    // head bias: waves 0..7 each own tile (mt = wave>>2, ntile = wave&3)
    float hbr = (wave < 8) ? head_b[(wave & 3) * 16 + nn] : 0.0f;

    // A-fragment address helper: element (m, k) of a 16x32 bf16 tile.
    //   lane = m + ((k>>3)&1)*16 ;  j = ((k>>4)&1)*8 + (k&7)
#define AST_IDX(mt, kt, la, j) (((((mt) * 32 + (kt)) * 32 + (la)) * 16) + (j))

    // ---- initial staging: x = embed[BOS], h = 0 ----
    for (int idx = tid; idx < TILE_B * 1024; idx += BLOCK) {
        int mrow = idx >> 10;                    // 0..31
        int k    = idx & 1023;
        float v = (k < HID) ? embed[(size_t)ALPH * HID + k] : 0.0f;  // BOS row
        int mt = mrow >> 4, m = mrow & 15;
        int kt = k >> 5, kk = k & 31;
        int la = m + ((kk >> 3) & 1) * 16;
        int j  = ((kk >> 4) & 1) * 8 + (kk & 7);
        Ast[AST_IDX(mt, kt, la, j)] = (__bf16)v;
    }
    __syncthreads();

    auto cell_gemm = [&]() {
        #pragma unroll
        for (int g = 0; g < 4; ++g)
            #pragma unroll
            for (int ct = 0; ct < 2; ++ct)
                #pragma unroll
                for (int mt = 0; mt < 2; ++mt) { v8f z = {}; acc[g][ct][mt] = z; }
        #pragma unroll 2
        for (int kt = 0; kt < 32; ++kt) {
            v16bf af0 = *(const v16bf*)(&Ast[AST_IDX(0, kt, lane, 0)]);
            v16bf af1 = *(const v16bf*)(&Ast[AST_IDX(1, kt, lane, 0)]);
            const __bf16* wb = Wc + (size_t)kt * 128 * 512 + (size_t)lane * 16;
            #pragma unroll
            for (int g = 0; g < 4; ++g) {
                #pragma unroll
                for (int ct = 0; ct < 2; ++ct) {
                    int ntile = g * 32 + wave * 2 + ct;
                    v16bf bf = *(const v16bf*)(wb + (size_t)ntile * 512);
                    acc[g][ct][0] = __builtin_amdgcn_wmma_f32_16x16x32_bf16(
                        false, af0, false, bf, (short)0, acc[g][ct][0], false, false);
                    acc[g][ct][1] = __builtin_amdgcn_wmma_f32_16x16x32_bf16(
                        false, af1, false, bf, (short)0, acc[g][ct][1], false, false);
                }
            }
        }
    };

    auto cell_elem = [&]() {
        #pragma unroll
        for (int ct = 0; ct < 2; ++ct) {
            int col = wave * 32 + ct * 16 + nn;       // 0..511 hidden column
            int kg  = HID + col;                      // K index of h region
            int kt  = kg >> 5, kk = kg & 31;
            int la_off = ((kk >> 3) & 1) * 16;
            int j      = ((kk >> 4) & 1) * 8 + (kk & 7);
            #pragma unroll
            for (int mt = 0; mt < 2; ++mt) {
                #pragma unroll
                for (int r = 0; r < 8; ++r) {
                    float iv = acc[0][ct][mt][r] + biasr[0][ct];
                    float fv = acc[1][ct][mt][r] + biasr[1][ct];
                    float gv = acc[2][ct][mt][r] + biasr[2][ct];
                    float ov = acc[3][ct][mt][r] + biasr[3][ct];
                    float cn = fsig(fv) * creg[ct][mt][r] + fsig(iv) * ftanh(gv);
                    creg[ct][mt][r] = cn;
                    float hn = fsig(ov) * ftanh(cn);
                    Ast[AST_IDX(mt, kt, (mbase + r) + la_off, j)] = (__bf16)hn;
                }
            }
        }
    };

    auto head_gemm = [&]() {
        if (wave < 8) {
            int mt    = wave >> 2;
            int ntile = wave & 3;
            v8f ha = {};
            #pragma unroll 2
            for (int kt = 0; kt < 16; ++kt) {
                v16bf af = *(const v16bf*)(&Ast[AST_IDX(mt, 16 + kt, lane, 0)]);
                v16bf bf = *(const v16bf*)(Hw + (size_t)(kt * 4 + ntile) * 512
                                              + (size_t)lane * 16);
                ha = __builtin_amdgcn_wmma_f32_16x16x32_bf16(
                    false, af, false, bf, (short)0, ha, false, false);
            }
            int n = ntile * 16 + nn;
            #pragma unroll
            for (int r = 0; r < 8; ++r)
                logits_s[(mt * 16 + mbase + r) * ALPH + n] = ha[r] + hbr;
        }
    };

    // BOS step -> logits for t=0
    cell_gemm(); __syncthreads();
    cell_elem(); __syncthreads();
    head_gemm(); __syncthreads();

    float logp = 0.0f;
    const int rowG = blockIdx.x * TILE_B + tid;   // valid for tid < TILE_B

    for (int t = 0; t < SEQL; ++t) {
        // ---- sample (Gumbel-max) + log-prob, one row per thread 0..31 ----
        if (tid < TILE_B) {
            const float* lg = &logits_s[tid * ALPH];
            float mx = -3.0e38f;
            for (int j2 = 0; j2 < ALPH; ++j2)
                mx = fmaxf(mx, lg[j2]);
            float se = 0.0f;
            for (int j2 = 0; j2 < ALPH; ++j2)
                se += __expf(lg[j2] - mx);
            float lse = __logf(se) + mx;

            uint32_t base = ((uint32_t)t * (uint32_t)BATCHN + (uint32_t)rowG) * 64u;
            float best = -3.0e38f; int bi = 0;
            for (int j2 = 0; j2 < ALPH; ++j2) {
                uint32_t rnd = mix32(base + (uint32_t)j2 + 0x9e3779b9u);
                float u = (float)(rnd >> 8) * 5.9604645e-08f + 1.0e-9f;
                float gmb = -__logf(-__logf(u));
                float v = lg[j2] + gmb;
                if (v > best) { best = v; bi = j2; }
            }
            logp += lg[bi] - lse;
            toks[tid] = bi;
            out_x[(size_t)rowG * SEQL + t] = bi;
        }
        __syncthreads();
        if (t == SEQL - 1) break;

        // ---- stage x = embed[tok] into A-fragments (bf16) ----
        for (int idx = tid; idx < TILE_B * HID; idx += BLOCK) {
            int mrow = idx >> 9;                 // 0..31
            int col  = idx & 511;
            int tok  = toks[mrow];
            float v = embed[(size_t)tok * HID + col];
            int mt = mrow >> 4, m = mrow & 15;
            int kt = col >> 5, kk = col & 31;
            int la = m + ((kk >> 3) & 1) * 16;
            int j  = ((kk >> 4) & 1) * 8 + (kk & 7);
            Ast[AST_IDX(mt, kt, la, j)] = (__bf16)v;
        }
        __syncthreads();

        cell_gemm(); __syncthreads();
        cell_elem(); __syncthreads();
        head_gemm(); __syncthreads();
    }

    if (tid < TILE_B) out_logp[rowG] = logp;
#undef AST_IDX
}

// ---------------------------------------------------------------------------
// Host launcher
// ---------------------------------------------------------------------------
extern "C" void kernel_launch(void* const* d_in, const int* in_sizes, int n_in,
                              void* d_out, int out_size, void* d_ws, size_t ws_size,
                              hipStream_t stream)
{
    (void)in_sizes; (void)n_in; (void)out_size; (void)ws_size;
    const float* embed  = (const float*)d_in[1];
    const float* w_ih   = (const float*)d_in[2];
    const float* w_hh   = (const float*)d_in[3];
    const float* b_ih   = (const float*)d_in[4];
    const float* b_hh   = (const float*)d_in[5];
    const float* head_w = (const float*)d_in[6];
    const float* head_b = (const float*)d_in[7];

    // workspace carve-out (bf16 weight fragments + combined bias)
    unsigned short* wc_raw = (unsigned short*)d_ws;                       // 4 MB
    unsigned short* hw_raw = (unsigned short*)((char*)d_ws + 4194304);    // 64 KB
    float*          bc     = (float*)((char*)d_ws + 4194304 + 65536);     // 8 KB

    int*   out_x    = (int*)d_out;
    float* out_logp = (float*)d_out + (size_t)BATCHN * SEQL;

    conv_wc_kernel<<<(1024 * 2048) / 256, 256, 0, stream>>>(w_ih, w_hh, wc_raw);
    conv_hw_kernel<<<(HID * ALPH) / 256, 256, 0, stream>>>(head_w, hw_raw);
    conv_bias_kernel<<<(4 * HID + 255) / 256, 256, 0, stream>>>(b_ih, b_hh, bc);

    lstm_sample_kernel<<<BATCHN / TILE_B, BLOCK, SMEM_TOTAL, stream>>>(
        embed, wc_raw, hw_raw, bc, head_b, out_x, out_logp);
}
